// MultiHeadAttention_14508399526320
// MI455X (gfx1250) — matrix operand order repaired and checked by reading, not compiled, not run on hardware
//
#include <hip/hip_runtime.h>
#include <hip/hip_bf16.h>

// ---------------------------------------------------------------------------
// MHA forward for MI455X (gfx1250, wave32, WMMA 16x16x32 f16 -> f32 acc)
// B=2, S=2048, D=1024, H=16, DK=64
// ---------------------------------------------------------------------------

#define MB 2
#define MS 2048
#define MD 1024
#define MH 16
#define MDK 64

typedef __attribute__((ext_vector_type(16))) _Float16 v16h;
typedef __attribute__((ext_vector_type(4)))  _Float16 v4h;
typedef __attribute__((ext_vector_type(8)))  float    v8f;

__device__ __forceinline__ v8f wmma32(v16h a, v16h b, v8f c) {
    // (neg_a, A, neg_b, B, c_mod, C, reuse_a, reuse_b)
    return __builtin_amdgcn_wmma_f32_16x16x32_f16(
        false, a, false, b, (short)0, c, false, false);
}

// A-operand (16x32 f16, row major, row stride ldk halfs). contiguous-k layout (B packed the same way):
// m = lane&15, half = lane>>4, VGPR j holds K = half*16+2*j .. +1
// Per lane this is one contiguous 32B run.
__device__ __forceinline__ v16h load_A_h16(const _Float16* base, int ldk, int lane) {
    const int m  = lane & 15;
    const int hh = lane >> 4;
    v16h a;
#pragma unroll
    for (int j = 0; j < 8; ++j) {
        const int k0 = hh * 16 + 2 * j;
        a[2 * j]     = base[(size_t)m * ldk + k0];
        a[2 * j + 1] = base[(size_t)m * ldk + k0 + 1];
    }
    return a;
}

// Same A layout, fp32 source in LDS converted on the fly.
__device__ __forceinline__ v16h load_A_f32(const float* base, int ldk, int lane) {
    const int m  = lane & 15;
    const int hh = lane >> 4;
    v16h a;
#pragma unroll
    for (int j = 0; j < 8; ++j) {
        const int k0 = hh * 16 + 2 * j;
        a[2 * j]     = (_Float16)base[(size_t)m * ldk + k0];
        a[2 * j + 1] = (_Float16)base[(size_t)m * ldk + k0 + 1];
    }
    return a;
}

// Packed B-operand layout index for V[b, s, h, d] (B-tile = 32x16, K=s, N=d):
// tile = ((b*H+h)*(S/32) + s/32)*4 + d/16 ; within tile lane = (s%32/16)*16 + d%16,
// element = lane*16 + 2*((s%16)/2) + (s&1)
__device__ __forceinline__ size_t packedV_index(int bi, int h, int s, int d) {
    const int kt  = s >> 5;
    const int w32 = s & 31;
    const int hh2 = w32 >> 4;
    const int r   = w32 & 15;
    const int dt  = d >> 4;
    const int n   = d & 15;
    return ((((size_t)(bi * MH + h) * (MS / 32) + kt) * 4 + dt) * 32 + hh2 * 16 + n) * 16
           + 2 * (r >> 1) + (r & 1);
}

// ---------------------------------------------------------------------------
// Kernel 1: convert q/k/v fp32 -> f16 (vectorized: float4 in, 4 halfs out)
// ---------------------------------------------------------------------------
__global__ void mha_cvt3(const float* __restrict__ q, const float* __restrict__ k,
                         const float* __restrict__ v,
                         _Float16* __restrict__ qo, _Float16* __restrict__ ko,
                         _Float16* __restrict__ vo, size_t n4) {
    size_t i4 = (size_t)blockIdx.x * blockDim.x + threadIdx.x;
    if (i4 >= n4) return;
    size_t i = i4 * 4;
    float4 xq = *(const float4*)(q + i);
    float4 xk = *(const float4*)(k + i);
    float4 xv = *(const float4*)(v + i);
    v4h hq = { (_Float16)xq.x, (_Float16)xq.y, (_Float16)xq.z, (_Float16)xq.w };
    v4h hk = { (_Float16)xk.x, (_Float16)xk.y, (_Float16)xk.z, (_Float16)xk.w };
    v4h hv = { (_Float16)xv.x, (_Float16)xv.y, (_Float16)xv.z, (_Float16)xv.w };
    *(v4h*)(qo + i) = hq;
    *(v4h*)(ko + i) = hk;
    *(v4h*)(vo + i) = hv;
}

// ---------------------------------------------------------------------------
// Kernel 2: pack weight [K=1024, N=1024] fp32 -> f16 WMMA B-operand tiles:
// tile t = kt*(N/16)+nt holds 32 lanes x 16 contiguous halfs.
// ---------------------------------------------------------------------------
__global__ void mha_pack_w(const float* __restrict__ W, _Float16* __restrict__ Wp,
                           int K, int N) {
    size_t idx = (size_t)blockIdx.x * blockDim.x + threadIdx.x;   // pair index
    size_t npairs = (size_t)K * N / 2;
    if (idx >= npairs) return;
    const int    j    = (int)(idx & 7);
    const int    lane = (int)((idx >> 3) & 31);
    const size_t tile = idx >> 8;                 // 256 pairs per 32x16 tile
    const int nt = (int)(tile % (N / 16));
    const int kt = (int)(tile / (N / 16));
    const int n  = nt * 16 + (lane & 15);
    const int k  = kt * 32 + (lane >> 4) * 16 + 2 * j;
    Wp[idx * 2]     = (_Float16)W[(size_t)k * N + n];
    Wp[idx * 2 + 1] = (_Float16)W[(size_t)(k + 1) * N + n];
}

// ---------------------------------------------------------------------------
// Kernel 3: GEMM  C[M,N] = X[M,K] @ Wp + bias.
// Block = 128 threads = 4 waves; wave owns a 32x64 tile (2 m-tiles x 4 n-tiles,
// 8 WMMAs per k-step) with software-pipelined (double-buffered) loads.
// mode: 0 = f16 row-major, 1 = f32 row-major, 2 = f16 packed-V layout
// ---------------------------------------------------------------------------
__global__ void mha_gemm(const _Float16* __restrict__ X, const _Float16* __restrict__ Wp,
                         const float* __restrict__ bias,
                         void* __restrict__ outp, int mode, int M, int N, int K) {
    const int lane  = threadIdx.x & 31;
    const int wv    = threadIdx.x >> 5;
    const int mwave = blockIdx.x * 4 + wv;        // 32-row strip index
    const int nb    = blockIdx.y * 64;
    const int KS    = K / 32;

    const _Float16* x0 = X + (size_t)mwave * 32 * K;
    const _Float16* x1 = x0 + (size_t)16 * K;
    const size_t kstride = (size_t)(N / 16) * 512;               // halfs per k-step
    const _Float16* wp0 = Wp + (((size_t)(nb >> 4)) * 32 + lane) * 16;

    v8f acc[2][4] = {};

    // prologue: k-step 0
    v16h a0 = load_A_h16(x0, K, lane);
    v16h a1 = load_A_h16(x1, K, lane);
    v16h b0 = *(const v16h*)(wp0);
    v16h b1 = *(const v16h*)(wp0 + 512);
    v16h b2 = *(const v16h*)(wp0 + 1024);
    v16h b3 = *(const v16h*)(wp0 + 1536);

    for (int ks = 0; ks < KS; ++ks) {
        const int ksn = (ks + 1 < KS) ? ks + 1 : ks;
        // issue next k-step's loads before the WMMAs so they overlap compute
        v16h a0n = load_A_h16(x0 + ksn * 32, K, lane);
        v16h a1n = load_A_h16(x1 + ksn * 32, K, lane);
        const _Float16* wn = wp0 + (size_t)ksn * kstride;
        v16h b0n = *(const v16h*)(wn);
        v16h b1n = *(const v16h*)(wn + 512);
        v16h b2n = *(const v16h*)(wn + 1024);
        v16h b3n = *(const v16h*)(wn + 1536);

        acc[0][0] = wmma32(a0, b0, acc[0][0]);
        acc[0][1] = wmma32(a0, b1, acc[0][1]);
        acc[0][2] = wmma32(a0, b2, acc[0][2]);
        acc[0][3] = wmma32(a0, b3, acc[0][3]);
        acc[1][0] = wmma32(a1, b0, acc[1][0]);
        acc[1][1] = wmma32(a1, b1, acc[1][1]);
        acc[1][2] = wmma32(a1, b2, acc[1][2]);
        acc[1][3] = wmma32(a1, b3, acc[1][3]);

        a0 = a0n; a1 = a1n; b0 = b0n; b1 = b1n; b2 = b2n; b3 = b3n;
    }

    const int n0 = lane & 15;
    const int hh = lane >> 4;
#pragma unroll
    for (int mi = 0; mi < 2; ++mi) {
#pragma unroll
        for (int nt = 0; nt < 4; ++nt) {
            const int col = nb + nt * 16 + n0;
            const float bval = bias[col];
#pragma unroll
            for (int r = 0; r < 8; ++r) {
                const int row = mwave * 32 + mi * 16 + r + 8 * hh;
                const float val = acc[mi][nt][r] + bval;
                if (mode == 0) {
                    ((_Float16*)outp)[(size_t)row * N + col] = (_Float16)val;
                } else if (mode == 1) {
                    ((float*)outp)[(size_t)row * N + col] = val;
                } else {
                    const int bi = row >> 11;          // row / MS
                    const int s  = row & (MS - 1);
                    const int h  = col >> 6;
                    const int d  = col & 63;
                    ((_Float16*)outp)[packedV_index(bi, h, s, d)] = (_Float16)val;
                }
            }
        }
    }
}

// ---------------------------------------------------------------------------
// Kernel 4: fused attention for one (b,h,16-row q-tile).
// Block = 128 threads = 4 waves. Dynamic LDS:
//   s_scores : 16 x 2048 fp32 (128 KB), s_q : 16x64 f16, reductions : 1 KB
// ---------------------------------------------------------------------------
__global__ void mha_attn_fused(const _Float16* __restrict__ Qf,
                               const _Float16* __restrict__ Kf,
                               const _Float16* __restrict__ Vpk,
                               float* __restrict__ attn_out,
                               _Float16* __restrict__ head_out) {
    const int bh = blockIdx.x;           // 0..B*H-1
    const int bi = bh / MH;
    const int hi = bh % MH;
    const int qt = blockIdx.y;           // 0..S/16-1
    const int tid  = threadIdx.x;
    const int lane = tid & 31;
    const int wv   = tid >> 5;

    extern __shared__ char smem_raw[];
    float*    s_scores = (float*)smem_raw;                       // 16*2048
    _Float16* s_q      = (_Float16*)(s_scores + 16 * MS);        // 16*64
    float*    s_rmax   = (float*)(s_q + 16 * MDK);               // 128
    float*    s_rsum   = s_rmax + 128;                           // 128

    // --- stage Q tile (2048 B) via async copy: 128 threads x b128 ---
    {
        const int i0 = tid * 8;                       // halfs
        const int qr = i0 >> 6;
        const int qc = i0 & 63;
        const _Float16* src =
            Qf + ((size_t)(bi * MS + qt * 16 + qr) * MD + hi * MDK + qc);
        unsigned lds_off = (unsigned)(uintptr_t)(void*)s_q + (unsigned)(tid * 16);
        asm volatile("global_load_async_to_lds_b128 %0, %1, off"
                     :: "v"(lds_off), "v"(src)
                     : "memory");
        asm volatile("s_wait_asynccnt 0x0" ::: "memory");
    }
    __syncthreads();

    // --- phase 1: scores = (Q K^T) * 1/sqrt(dk); waves stride k-tiles ---
    const float scale = 0.125f;  // 1/sqrt(64)
    for (int kt = wv; kt < MS / 16; kt += 4) {
        // lane owns key row n = kt*16 + (lane&15); contiguous 32B per chunk
        const _Float16* krow =
            Kf + ((size_t)(bi * MS + kt * 16 + (lane & 15)) * MD + hi * MDK);
        if (kt + 4 < MS / 16)
            __builtin_prefetch(krow + (size_t)64 * MD, 0, 0);
        v8f acc = {};
#pragma unroll
        for (int ch = 0; ch < 2; ++ch) {
            v16h a = load_A_h16(s_q + ch * 32, MDK, lane);
            v16h b = *(const v16h*)(krow + ch * 32 + (lane >> 4) * 16);
            acc = wmma32(a, b, acc);
        }
        const int n  = lane & 15;
        const int hh = lane >> 4;
#pragma unroll
        for (int r = 0; r < 8; ++r) {
            const int m = r + 8 * hh;
            s_scores[(size_t)m * MS + kt * 16 + n] = acc[r] * scale;
        }
    }
    __syncthreads();

    // --- phase 2: row softmax (8 threads/row, float4), fp32 attn to d_out ---
    {
        const int row = tid >> 3;    // 0..15
        const int sub = tid & 7;     // 0..7
        float* srow = s_scores + (size_t)row * MS;

        float lmax = -3.4e38f;
        for (int c = sub * 4; c < MS; c += 32) {
            float4 x = *(const float4*)(srow + c);
            lmax = fmaxf(lmax, fmaxf(fmaxf(x.x, x.y), fmaxf(x.z, x.w)));
        }
        s_rmax[row * 8 + sub] = lmax;
        __syncthreads();
        float rmax = s_rmax[row * 8];
#pragma unroll
        for (int i = 1; i < 8; ++i) rmax = fmaxf(rmax, s_rmax[row * 8 + i]);

        float lsum = 0.f;
        for (int c = sub * 4; c < MS; c += 32) {
            float4 x = *(const float4*)(srow + c);
            x.x = __expf(x.x - rmax);
            x.y = __expf(x.y - rmax);
            x.z = __expf(x.z - rmax);
            x.w = __expf(x.w - rmax);
            *(float4*)(srow + c) = x;
            lsum += (x.x + x.y) + (x.z + x.w);
        }
        s_rsum[row * 8 + sub] = lsum;
        __syncthreads();
        float rsum = 0.f;
#pragma unroll
        for (int i = 0; i < 8; ++i) rsum += s_rsum[row * 8 + i];
        const float inv = 1.0f / rsum;

        float* grow = attn_out + ((size_t)bh * MS + qt * 16 + row) * MS;
        for (int c = sub * 4; c < MS; c += 32) {
            float4 x = *(const float4*)(srow + c);
            x.x *= inv; x.y *= inv; x.z *= inv; x.w *= inv;
            *(float4*)(srow + c) = x;
            *(float4*)(grow + c) = x;    // fp32 attn output (roofline term)
        }
    }
    __syncthreads();

    // --- phase 3: head_out = attn @ V ; wave wv owns d-tile wv (16 cols).
    //     V is pre-packed in B-operand layout -> contiguous 32B per lane.
    //     Double-buffered over the 64 k-chunks.
    {
        const _Float16* vbase =
            Vpk + (((size_t)bh * (MS / 32)) * 4 + wv) * 512 + (size_t)lane * 16;
        v8f acc = {};
        v16h a = load_A_f32(s_scores, MS, lane);
        v16h b = *(const v16h*)(vbase);
        for (int kt = 0; kt < MS / 32; ++kt) {
            const int ktn = (kt + 1 < MS / 32) ? kt + 1 : kt;
            v16h an = load_A_f32(s_scores + ktn * 32, MS, lane);
            v16h bn = *(const v16h*)(vbase + (size_t)ktn * 2048);
            acc = wmma32(a, b, acc);
            a = an; b = bn;
        }
        const int n  = lane & 15;
        const int hh = lane >> 4;
#pragma unroll
        for (int r = 0; r < 8; ++r) {
            const int row = qt * 16 + r + 8 * hh;
            head_out[(size_t)(bi * MS + row) * MD + hi * MDK + wv * 16 + n] =
                (_Float16)acc[r];
        }
    }
}

// ---------------------------------------------------------------------------
// Host launcher
// ---------------------------------------------------------------------------
extern "C" void kernel_launch(void* const* d_in, const int* in_sizes, int n_in,
                              void* d_out, int out_size, void* d_ws, size_t ws_size,
                              hipStream_t stream) {
    (void)in_sizes; (void)n_in; (void)out_size; (void)ws_size;

    const float* q_in = (const float*)d_in[0];
    const float* k_in = (const float*)d_in[1];
    const float* v_in = (const float*)d_in[2];
    const float* Wq = (const float*)d_in[3];  const float* bq = (const float*)d_in[4];
    const float* Wk = (const float*)d_in[5];  const float* bk = (const float*)d_in[6];
    const float* Wv = (const float*)d_in[7];  const float* bv = (const float*)d_in[8];
    const float* Wo = (const float*)d_in[9];  const float* bo = (const float*)d_in[10];

    const size_t NE = (size_t)MB * MS * MD;        // 4,194,304 activation elems
    const size_t WN = (size_t)MD * MD;             // 1,048,576 weight elems

    _Float16* ws = (_Float16*)d_ws;                // all offsets in halfs
    _Float16* qf16 = ws + 0 * NE;                  // converted inputs
    _Float16* kf16 = ws + 1 * NE;
    _Float16* vf16 = ws + 2 * NE;
    _Float16* Qp   = ws + 3 * NE;                  // projected Q (f16 row-major)
    _Float16* Kp   = ws + 4 * NE;                  // projected K (f16 row-major)
    _Float16* Vpk  = ws + 5 * NE;                  // projected V (f16 B-packed)
    _Float16* AO   = ws + 6 * NE;                  // concat head outputs (f16)
    _Float16* wq_p = ws + 7 * NE;                  // packed f16 weights
    _Float16* wk_p = wq_p + WN;
    _Float16* wv_p = wk_p + WN;
    _Float16* wo_p = wv_p + WN;

    float* out_f  = (float*)d_out;                 // [B,S,D]
    float* attn_f = out_f + NE;                    // [B,H,S,S]

    // 1) fp32 -> f16 inputs
    mha_cvt3<<<dim3((unsigned)(NE / 4 / 256)), dim3(256), 0, stream>>>(
        q_in, k_in, v_in, qf16, kf16, vf16, NE / 4);

    // 2) pack weights into WMMA B-operand layout
    {
        const size_t pairs = WN / 2;
        dim3 g((unsigned)((pairs + 255) / 256)), t(256);
        mha_pack_w<<<g, t, 0, stream>>>(Wq, wq_p, MD, MD);
        mha_pack_w<<<g, t, 0, stream>>>(Wk, wk_p, MD, MD);
        mha_pack_w<<<g, t, 0, stream>>>(Wv, wv_p, MD, MD);
        mha_pack_w<<<g, t, 0, stream>>>(Wo, wo_p, MD, MD);
    }

    // 3) Q/K/V projections: [4096 x 1024] @ [1024 x 1024] + b
    {
        const int M = MB * MS;
        dim3 g(M / 128, MD / 64), t(128);
        mha_gemm<<<g, t, 0, stream>>>(qf16, wq_p, bq, Qp,  0, M, MD, MD);
        mha_gemm<<<g, t, 0, stream>>>(kf16, wk_p, bk, Kp,  0, M, MD, MD);
        mha_gemm<<<g, t, 0, stream>>>(vf16, wv_p, bv, Vpk, 2, M, MD, MD);
    }

    // 4) fused scores + softmax + attn@V (one block per (b,h,16-row q-tile))
    {
        const size_t lds = 16 * MS * sizeof(float)       // scores row-block
                         + 16 * MDK * sizeof(_Float16)   // Q tile
                         + 256 * sizeof(float);          // reductions
        hipFuncSetAttribute((const void*)mha_attn_fused,
                            hipFuncAttributeMaxDynamicSharedMemorySize, (int)lds);
        dim3 g(MB * MH, MS / 16), t(128);
        mha_attn_fused<<<g, t, lds, stream>>>(Qp, Kp, Vpk, attn_f, AO);
    }

    // 5) output projection -> fp32 d_out
    {
        const int M = MB * MS;
        dim3 g(M / 128, MD / 64), t(128);
        mha_gemm<<<g, t, 0, stream>>>(AO, wo_p, bo, out_f, 1, M, MD, MD);
    }
}
